// HeterogeneousGraphTransformer_17076789969452
// MI455X (gfx1250) — compile-verified, matrix operand order checked by reading, not run on hardware
//
#include <hip/hip_runtime.h>
#include <hip/hip_bf16.h>
#include <cstring>

typedef __bf16 bf16;
typedef __attribute__((ext_vector_type(16))) __bf16 v16bf;
typedef __attribute__((ext_vector_type(8)))  float  v8f;
typedef __attribute__((ext_vector_type(4)))  int    v4i;
typedef __attribute__((ext_vector_type(8)))  int    v8i;

#define LDIM  3
#define HN    8
#define HDQ   32
#define DMODEL 256
#define TNN   4
#define TEE   3

// ---------------- WMMA helpers (CDNA5 16x16x32 bf16 -> f32) ----------------

__device__ inline v8f wmma_bf(v16bf a, v16bf b, v8f c) {
  return __builtin_amdgcn_wmma_f32_16x16x32_bf16(false, a, false, b, (short)0, c,
                                                 false, false);
}

// 16-bit A-matrix 16x32 layout: K pair base for vgpr v, lane-half h.
__device__ inline int kpair_base(int v, int half) {
  return ((v & 4) << 2) + ((v & 3) << 1) + (half << 3);
}

// A tile (16 rows x 32 K) from pair-contiguous bf16 rows: per lane the data is
// elements [8h, 8h+8) and [16+8h, 16+8h+8) of the row -> two 16B vector loads
// (global_load_b128 or ds_load_b128 depending on the pointer).
__device__ inline v16bf load_a_bf16(const bf16* tile, int ld, int lane, int mclamp) {
  int half = lane >> 4;
  int m = lane & 15; if (m > mclamp) m = mclamp;
  const bf16* r = tile + (size_t)m * ld;
  v4i lo = *(const v4i*)(r + 8 * half);
  v4i hi = *(const v4i*)(r + 16 + 8 * half);
  v8i t;
  t[0] = lo[0]; t[1] = lo[1]; t[2] = lo[2]; t[3] = lo[3];
  t[4] = hi[0]; t[5] = hi[1]; t[6] = hi[2]; t[7] = hi[3];
  return __builtin_bit_cast(v16bf, t);
}

// B tile from pre-swizzled weights: lane's 16 bf16 are 32 contiguous bytes.
__device__ inline v16bf load_b_swz(const bf16* tilebase, int lane) {
  return __builtin_bit_cast(v16bf, *(const v8i*)(tilebase + lane * 16));
}

__device__ inline float wave_sum(float v) {
#pragma unroll
  for (int m = 16; m >= 1; m >>= 1) v += __shfl_xor(v, m, 32);
  return v;
}

// order-preserving float<->uint encoding for atomic max
__device__ inline unsigned encf(float f) {
  unsigned u = __float_as_uint(f);
  return (u & 0x80000000u) ? ~u : (u | 0x80000000u);
}
__device__ inline float decf(unsigned u) {
  unsigned b = (u & 0x80000000u) ? (u ^ 0x80000000u) : ~u;
  return __uint_as_float(b);
}

// ---------------- utility kernels ----------------

__global__ void cvt_bf16_kernel(const float* __restrict__ in, bf16* __restrict__ out, int n) {
  int t = blockIdx.x * blockDim.x + threadIdx.x;
  if (t < n) out[t] = (bf16)in[t];
}

__global__ void fill_u32_kernel(unsigned* __restrict__ p, unsigned val, int n) {
  int t = blockIdx.x * blockDim.x + threadIdx.x;
  if (t < n) p[t] = val;
}

// Convert f32 weights to bf16 AND swizzle into per-lane WMMA B layout.
// Matrix [K x Ncols] -> tiles [c (Ncols/16)][t (K/32)], each tile 512 bf16:
// tile[lane*16 + 2v+j] = B[kpair_base(v,half)+j][n], n=lane&15, half=lane>>4.
// transpose=1 reads W as column-major (used for W_E^T; square matrices only).
__global__ void __launch_bounds__(256)
swz_b_kernel(const float* __restrict__ W, bf16* __restrict__ out,
             int nmat, int K, int Ncols, int transpose) {
  int wave = (blockIdx.x * blockDim.x + threadIdx.x) >> 5;
  int lane = threadIdx.x & 31;
  int kt = K >> 5, nt = Ncols >> 4;
  int tilesPerMat = kt * nt;
  if (wave >= nmat * tilesPerMat) return;
  int mat = wave / tilesPerMat;
  int rem = wave - mat * tilesPerMat;            // rem = c*kt + t
  int c = rem / kt, t = rem - c * kt;
  const float* Wm = W + (size_t)mat * K * Ncols;
  bf16* om = out + (size_t)mat * K * Ncols + (size_t)rem * 512 + lane * 16;
  int half = lane >> 4, n = lane & 15;
#pragma unroll
  for (int v = 0; v < 8; ++v) {
    int kb = kpair_base(v, half);
    int k = t * 32 + kb, col = c * 16 + n;
    float a0, a1;
    if (transpose) { a0 = Wm[(size_t)col * K + k];     a1 = Wm[(size_t)col * K + k + 1]; }
    else           { a0 = Wm[(size_t)k * Ncols + col]; a1 = Wm[(size_t)(k + 1) * Ncols + col]; }
    om[2 * v]     = (bf16)a0;
    om[2 * v + 1] = (bf16)a1;
  }
}

// ---------------- fused typed QKV transform + qhat precompute ----------------
// One wave per (16-node tile, head). Per-type WMMA + per-row select (row types
// from broadcast loads). Q is staged in LDS as bf16 in pair-contiguous layout,
// reloaded via ds_load_b128 as the A operand for qhat[et] = W_E[et]^T * q
// (W_E pre-transposed in the swizzled weights).
__global__ void __launch_bounds__(32)
qkv_qhat_kernel(const bf16* __restrict__ xb, const int* __restrict__ node_type,
                const bf16* __restrict__ Wq, const bf16* __restrict__ Wk,
                const bf16* __restrict__ Wv, const bf16* __restrict__ We,
                float* __restrict__ Kf, float* __restrict__ Vf,
                float* __restrict__ qhat, int N) {
  __shared__ __align__(64) bf16 qs[16 * HDQ];
  int lane = threadIdx.x;
  int tile = blockIdx.x, h = blockIdx.y;
  int row0 = tile * 16;
  int mclamp = N - 1 - row0; if (mclamp > 15) mclamp = 15;
  int half = lane >> 4, n = lane & 15;

  // row type for each of this lane's 8 C-layout rows (M = vi + 8*half)
  int rt[8];
#pragma unroll
  for (int vi = 0; vi < 8; ++vi) {
    int r = row0 + vi + 8 * half; if (r > N - 1) r = N - 1;
    rt[vi] = node_type[r];
  }
  // presence ballot covers all 16 rows via each lane's own row
  int ownrow = row0 + (lane & 15); if (ownrow > N - 1) ownrow = N - 1;
  int mytype = node_type[ownrow];

  v16bf a = load_a_bf16(xb + (size_t)row0 * DMODEL + h * HDQ, DMODEL, lane, mclamp);

  v8f accQ0 = {}, accQ1 = {}, accK0 = {}, accK1 = {}, accV0 = {}, accV1 = {};
#pragma unroll
  for (int t = 0; t < TNN; ++t) {
    if (__ballot(mytype == t)) {            // wave-uniform: EXEC stays full for WMMA
      const bf16* wq = Wq + (size_t)(t * HN + h) * (HDQ * HDQ);
      const bf16* wk = Wk + (size_t)(t * HN + h) * (HDQ * HDQ);
      const bf16* wv = Wv + (size_t)(t * HN + h) * (HDQ * HDQ);
      v8f z = {};
      v8f yq0 = wmma_bf(a, load_b_swz(wq,       lane), z);
      v8f yq1 = wmma_bf(a, load_b_swz(wq + 512, lane), z);
      v8f yk0 = wmma_bf(a, load_b_swz(wk,       lane), z);
      v8f yk1 = wmma_bf(a, load_b_swz(wk + 512, lane), z);
      v8f yv0 = wmma_bf(a, load_b_swz(wv,       lane), z);
      v8f yv1 = wmma_bf(a, load_b_swz(wv + 512, lane), z);
#pragma unroll
      for (int vi = 0; vi < 8; ++vi) {
        bool sel = (rt[vi] == t);
        accQ0[vi] = sel ? yq0[vi] : accQ0[vi];
        accQ1[vi] = sel ? yq1[vi] : accQ1[vi];
        accK0[vi] = sel ? yk0[vi] : accK0[vi];
        accK1[vi] = sel ? yk1[vi] : accK1[vi];
        accV0[vi] = sel ? yv0[vi] : accV0[vi];
        accV1[vi] = sel ? yv1[vi] : accV1[vi];
      }
    }
  }

  // store K,V (f32 for the edge kernels); stage Q in LDS (bf16, A-layout rows)
#pragma unroll
  for (int vi = 0; vi < 8; ++vi) {
    int Mr = vi + 8 * half;
    int row = row0 + Mr;
    if (row < N) {
      Kf[(size_t)row * DMODEL + h * HDQ + n]      = accK0[vi];
      Kf[(size_t)row * DMODEL + h * HDQ + 16 + n] = accK1[vi];
      Vf[(size_t)row * DMODEL + h * HDQ + n]      = accV0[vi];
      Vf[(size_t)row * DMODEL + h * HDQ + 16 + n] = accV1[vi];
    }
    qs[Mr * HDQ + n]      = (bf16)accQ0[vi];
    qs[Mr * HDQ + 16 + n] = (bf16)accQ1[vi];
  }
  __syncthreads();

  v16bf aq = load_a_bf16(qs, HDQ, lane, 15);
#pragma unroll
  for (int et = 0; et < TEE; ++et) {
    const bf16* we = We + (size_t)(et * HN + h) * (HDQ * HDQ);  // pre-transposed
    v8f z = {};
    v8f y0 = wmma_bf(aq, load_b_swz(we,       lane), z);
    v8f y1 = wmma_bf(aq, load_b_swz(we + 512, lane), z);
#pragma unroll
    for (int vi = 0; vi < 8; ++vi) {
      int row = row0 + vi + 8 * half;
      if (row < N) {
        float* qr = qhat + (((size_t)row * TEE + et) * HN + h) * HDQ;
        qr[n]      = y0[vi];
        qr[16 + n] = y1[vi];
      }
    }
  }
}

// ---------------- per-edge scores: (W_E q_dst) . k_src ----------------
__global__ void __launch_bounds__(256)
score_kernel(const float* __restrict__ qhat, const float* __restrict__ Kf,
             const int* __restrict__ ei, const int* __restrict__ etype,
             const float* __restrict__ mu_l, float* __restrict__ scores,
             unsigned* __restrict__ mEnc, int E) {
  int lane = threadIdx.x & 31;
  int e = blockIdx.x * 8 + (threadIdx.x >> 5);
  if (e >= E) return;
  int s = ei[e], d = ei[E + e], et = etype[e];
  const float* kr = Kf + (size_t)s * DMODEL;
  const float* qr = qhat + ((size_t)d * TEE + et) * DMODEL;
#pragma unroll
  for (int h = 0; h < HN; ++h) {
    float v = qr[h * HDQ + lane] * kr[h * HDQ + lane];
    v = wave_sum(v);
    float sc = v * 0.17677669529663687f * mu_l[h * TEE + et];  // 1/sqrt(32)
    if (lane == 0) {
      scores[(size_t)e * HN + h] = sc;
      atomicMax(&mEnc[(size_t)d * HN + h], encf(sc));
    }
  }
}

__global__ void expsum_kernel(float* __restrict__ scores, const unsigned* __restrict__ mEnc,
                              float* __restrict__ ssum, const int* __restrict__ ei, int E) {
  int t = blockIdx.x * blockDim.x + threadIdx.x;
  if (t >= E * HN) return;
  int e = t / HN, h = t - e * HN;
  int d = ei[E + e];
  float m = decf(mEnc[(size_t)d * HN + h]);     // init = -1e9, matches reference clamp
  float w = expf(scores[t] - m);
  scores[t] = w;
  atomicAdd(&ssum[(size_t)d * HN + h], w);
}

__global__ void __launch_bounds__(256)
agg_kernel(const float* __restrict__ scores, const float* __restrict__ ssum,
           const float* __restrict__ Vf, const int* __restrict__ ei,
           float* __restrict__ agg, int E) {
  int lane = threadIdx.x & 31;
  int e = blockIdx.x * 8 + (threadIdx.x >> 5);
  if (e >= E) return;
  int s = ei[e], d = ei[E + e];
  const float* vr = Vf + (size_t)s * DMODEL;
  float* ar = agg + (size_t)d * DMODEL;
#pragma unroll
  for (int h = 0; h < HN; ++h) {
    float w = scores[(size_t)e * HN + h] / (ssum[(size_t)d * HN + h] + 1e-10f);
    atomicAdd(&ar[h * HDQ + lane], w * vr[h * HDQ + lane]);
  }
}

// ---------------- generic WMMA GEMM ----------------
// out_f32 = act(A @ W + bias) (+ resid); optional bf16 mirror of the output.
// A is bf16 row-major; W is bf16 pre-swizzled. grid.x=ceil(M/16), grid.y=Ncols/128.
// The 16-row A tile is staged into LDS ONCE per block via CDNA5 async copies
// (global_load_async_to_lds_b128, ASYNCcnt), so the VMEM pipe in the k-loop is
// dedicated to streaming the swizzled weights; A is re-read via ds_load_b128.
__global__ void __launch_bounds__(256)
gemm_kernel(const bf16* __restrict__ A, const bf16* __restrict__ W,
            const float* __restrict__ bias, const float* __restrict__ resid,
            float* __restrict__ out, bf16* __restrict__ outbf,
            int M, int K, int Ncols, int act) {
  extern __shared__ bf16 sA[];                     // 16 rows x K bf16
  int lane = threadIdx.x & 31, wave = threadIdx.x >> 5;
  int row0 = blockIdx.x * 16;
  int c = blockIdx.y * 8 + wave;                   // 16-wide column tile index
  int col0 = c * 16;
  int kt = K >> 5;
  int mclamp = M - 1 - row0; if (mclamp > 15) mclamp = 15;

  // ---- async stage A tile: 16 rows x 2K bytes, 16B chunks across the block
  {
    unsigned ldsBase = (unsigned)(uintptr_t)sA;    // LDS aperture: low 32 bits = LDS addr
    int chunksPerRow = K >> 3;                     // 16B chunks per 2K-byte row
    int totalChunks = chunksPerRow << 4;
    int bytesPerRow = K << 1;
    for (int ci = threadIdx.x; ci < totalChunks; ci += 256) {
      int m = ci / chunksPerRow;
      int off = (ci - m * chunksPerRow) << 4;
      int mm = m > mclamp ? mclamp : m;
      unsigned long long ga = (unsigned long long)(uintptr_t)A
                            + (unsigned long long)(row0 + mm) * bytesPerRow + off;
      unsigned la = ldsBase + (unsigned)(m * bytesPerRow + off);
      asm volatile("global_load_async_to_lds_b128 %0, %1, off"
                   :: "v"(la), "v"(ga) : "memory");
    }
    asm volatile("s_wait_asynccnt 0" ::: "memory");
    __syncthreads();
  }

  const bf16* Wc = W + (size_t)c * kt * 512;       // this column-tile's k-stream
  v8f acc = {};
  for (int t = 0; t < kt; ++t) {
    v16bf a = load_a_bf16(sA + t * 32, K, lane, 15);   // ds_load_b128 x2
    v16bf b = load_b_swz(Wc + (size_t)t * 512, lane);  // global_load_b128 x2
    if (t + 1 < kt)
      __builtin_prefetch(Wc + (size_t)(t + 1) * 512 + lane * 16, 0, 1);
    acc = wmma_bf(a, b, acc);
  }
  int half = lane >> 4, n = lane & 15;
  int col = col0 + n;
  float bv = bias[col];
#pragma unroll
  for (int vi = 0; vi < 8; ++vi) {
    int row = row0 + vi + 8 * half;
    if (row < M) {
      float v = acc[vi] + bv;
      if (act) v = 0.5f * v * (1.0f + erff(v * 0.70710678118654752f)); // exact gelu
      if (resid) v += resid[(size_t)row * Ncols + col];
      if (out)   out[(size_t)row * Ncols + col] = v;
      if (outbf) outbf[(size_t)row * Ncols + col] = (bf16)v;
    }
  }
}

// ---------------- LayerNorm: one wave per 256-wide row; f32 + bf16 outputs ---
__global__ void __launch_bounds__(256)
ln_kernel(const float* __restrict__ src, const float* __restrict__ g,
          const float* __restrict__ b, float* __restrict__ dst,
          bf16* __restrict__ dstb, int N) {
  int lane = threadIdx.x & 31;
  int node = blockIdx.x * 8 + (threadIdx.x >> 5);
  if (node >= N) return;
  const float* r = src + (size_t)node * DMODEL;
  float v[8], s = 0.f, sq = 0.f;
#pragma unroll
  for (int i = 0; i < 8; ++i) { v[i] = r[i * 32 + lane]; s += v[i]; sq += v[i] * v[i]; }
  s = wave_sum(s); sq = wave_sum(sq);
  float mean = s * (1.0f / DMODEL);
  float var  = sq * (1.0f / DMODEL) - mean * mean;
  float inv  = rsqrtf(var + 1e-5f);
#pragma unroll
  for (int i = 0; i < 8; ++i) {
    int cidx = i * 32 + lane;
    float o = (v[i] - mean) * inv * g[cidx] + b[cidx];
    dst[(size_t)node * DMODEL + cidx] = o;
    if (dstb) dstb[(size_t)node * DMODEL + cidx] = (bf16)o;
  }
}

// ---------------- host side ----------------

static unsigned encf_host(float f) {
  unsigned u; std::memcpy(&u, &f, 4);
  return (u & 0x80000000u) ? ~u : (u | 0x80000000u);
}

extern "C" void kernel_launch(void* const* d_in, const int* in_sizes, int n_in,
                              void* d_out, int out_size, void* d_ws, size_t ws_size,
                              hipStream_t stream) {
  const float* x_in  = (const float*)d_in[0];
  const int*   ei    = (const int*)d_in[1];
  const int*   etype = (const int*)d_in[2];
  const int*   ntype = (const int*)d_in[3];
  const float* W_Q   = (const float*)d_in[4];
  const float* W_K   = (const float*)d_in[5];
  const float* W_V   = (const float*)d_in[6];
  const float* W_E   = (const float*)d_in[7];
  const float* mu    = (const float*)d_in[8];
  const float* Wo    = (const float*)d_in[9];
  const float* bo    = (const float*)d_in[10];
  const float* ln1g  = (const float*)d_in[11];
  const float* ln1b  = (const float*)d_in[12];
  const float* ln2g  = (const float*)d_in[13];
  const float* ln2b  = (const float*)d_in[14];
  const float* w1    = (const float*)d_in[15];
  const float* b1    = (const float*)d_in[16];
  const float* w2    = (const float*)d_in[17];
  const float* b2    = (const float*)d_in[18];
  const float* outg  = (const float*)d_in[19];
  const float* outb  = (const float*)d_in[20];

  const int N = in_sizes[3];
  const int E = in_sizes[2];

  // workspace carve (256B aligned)
  char* p = (char*)d_ws;
  auto carve = [&](size_t bytes) -> void* {
    char* r = p; p += (bytes + 255) & ~(size_t)255; return (void*)r;
  };
  float*    xA     = (float*)carve((size_t)N * DMODEL * 4);
  float*    xB     = (float*)carve((size_t)N * DMODEL * 4);
  bf16*     xb     = (bf16*)carve((size_t)N * DMODEL * 2);
  float*    Kf     = (float*)carve((size_t)N * DMODEL * 4);
  float*    Vf     = (float*)carve((size_t)N * DMODEL * 4);
  float*    agg    = (float*)carve((size_t)N * DMODEL * 4);
  bf16*     aggb   = (bf16*)carve((size_t)N * DMODEL * 2);
  bf16*     hb     = (bf16*)carve((size_t)N * 4 * DMODEL * 2);
  float*    qhat   = (float*)carve((size_t)N * TEE * DMODEL * 4);
  float*    scores = (float*)carve((size_t)E * HN * 4);
  unsigned* mEnc   = (unsigned*)carve((size_t)N * HN * 4);
  float*    ssum   = (float*)carve((size_t)N * HN * 4);
  bf16* Wqb = (bf16*)carve((size_t)LDIM * TNN * HN * HDQ * HDQ * 2);
  bf16* Wkb = (bf16*)carve((size_t)LDIM * TNN * HN * HDQ * HDQ * 2);
  bf16* Wvb = (bf16*)carve((size_t)LDIM * TNN * HN * HDQ * HDQ * 2);
  bf16* Web = (bf16*)carve((size_t)LDIM * TEE * HN * HDQ * HDQ * 2);
  bf16* Wob = (bf16*)carve((size_t)LDIM * DMODEL * DMODEL * 2);
  bf16* w1b = (bf16*)carve((size_t)LDIM * DMODEL * 4 * DMODEL * 2);
  bf16* w2b = (bf16*)carve((size_t)LDIM * 4 * DMODEL * DMODEL * 2);

  // weight convert + swizzle into WMMA B layout
  auto swz = [&](const float* src, bf16* dst, int nmat, int K, int Ncols, int tr) {
    int waves = nmat * (K >> 5) * (Ncols >> 4);
    swz_b_kernel<<<(waves * 32 + 255) / 256, 256, 0, stream>>>(src, dst, nmat, K, Ncols, tr);
  };
  swz(W_Q, Wqb, LDIM * TNN * HN, HDQ, HDQ, 0);
  swz(W_K, Wkb, LDIM * TNN * HN, HDQ, HDQ, 0);
  swz(W_V, Wvb, LDIM * TNN * HN, HDQ, HDQ, 0);
  swz(W_E, Web, LDIM * TEE * HN, HDQ, HDQ, 1);   // pre-transposed for qhat
  swz(Wo,  Wob, LDIM, DMODEL, DMODEL, 0);
  swz(w1,  w1b, LDIM, DMODEL, 4 * DMODEL, 0);
  swz(w2,  w2b, LDIM, 4 * DMODEL, DMODEL, 0);

  hipMemcpyAsync(xA, x_in, (size_t)N * DMODEL * 4, hipMemcpyDeviceToDevice, stream);
  cvt_bf16_kernel<<<(N * DMODEL + 255) / 256, 256, 0, stream>>>(x_in, xb, N * DMODEL);

  const unsigned encNeg = encf_host(-1e9f);
  const int tiles = (N + 15) / 16;
  const int ewaves = (E + 7) / 8;

  for (int l = 0; l < LDIM; ++l) {
    fill_u32_kernel<<<(N * HN + 255) / 256, 256, 0, stream>>>(mEnc, encNeg, N * HN);
    fill_u32_kernel<<<(N * HN + 255) / 256, 256, 0, stream>>>((unsigned*)ssum, 0u, N * HN);
    fill_u32_kernel<<<(N * DMODEL + 255) / 256, 256, 0, stream>>>((unsigned*)agg, 0u, N * DMODEL);

    qkv_qhat_kernel<<<dim3(tiles, HN), 32, 0, stream>>>(
        xb, ntype,
        Wqb + (size_t)l * TNN * HN * HDQ * HDQ,
        Wkb + (size_t)l * TNN * HN * HDQ * HDQ,
        Wvb + (size_t)l * TNN * HN * HDQ * HDQ,
        Web + (size_t)l * TEE * HN * HDQ * HDQ,
        Kf, Vf, qhat, N);

    score_kernel<<<ewaves, 256, 0, stream>>>(qhat, Kf, ei, etype,
                                             mu + (size_t)l * HN * TEE, scores, mEnc, E);
    expsum_kernel<<<(E * HN + 255) / 256, 256, 0, stream>>>(scores, mEnc, ssum, ei, E);
    agg_kernel<<<ewaves, 256, 0, stream>>>(scores, ssum, Vf, ei, agg, E);
    cvt_bf16_kernel<<<(N * DMODEL + 255) / 256, 256, 0, stream>>>(agg, aggb, N * DMODEL);

    // attn out projection + residual   (LDS = 16 rows * K bf16 = 32*K bytes)
    gemm_kernel<<<dim3(tiles, DMODEL / 128), 256, 32 * DMODEL, stream>>>(
        aggb, Wob + (size_t)l * DMODEL * DMODEL, bo + (size_t)l * DMODEL,
        xA, xB, nullptr, N, DMODEL, DMODEL, 0);
    ln_kernel<<<(N + 7) / 8, 256, 0, stream>>>(xB, ln1g + (size_t)l * DMODEL,
                                               ln1b + (size_t)l * DMODEL, xA, xb, N);
    // FFN (hidden lives only in bf16)
    gemm_kernel<<<dim3(tiles, (4 * DMODEL) / 128), 256, 32 * DMODEL, stream>>>(
        xb, w1b + (size_t)l * DMODEL * 4 * DMODEL, b1 + (size_t)l * 4 * DMODEL,
        nullptr, nullptr, hb, N, DMODEL, 4 * DMODEL, 1);
    gemm_kernel<<<dim3(tiles, DMODEL / 128), 256, 32 * 4 * DMODEL, stream>>>(
        hb, w2b + (size_t)l * 4 * DMODEL * DMODEL, b2 + (size_t)l * DMODEL,
        xA, xB, nullptr, N, 4 * DMODEL, DMODEL, 0);
    ln_kernel<<<(N + 7) / 8, 256, 0, stream>>>(xB, ln2g + (size_t)l * DMODEL,
                                               ln2b + (size_t)l * DMODEL, xA, xb, N);
  }

  ln_kernel<<<(N + 7) / 8, 256, 0, stream>>>(xA, outg, outb, (float*)d_out, nullptr, N);
}
